// NaiveMatching_7842610282738
// MI455X (gfx1250) — matrix operand order repaired and checked by reading, not compiled, not run on hardware
//
#include <hip/hip_runtime.h>
#include <hip/hip_bf16.h>
#include <math.h>

// Problem constants (B=1, C=3, H=192, W=320, KSIZE=13)
#define IMH 192
#define IMW 320
#define IMC 3
#define KS  13
#define PADK 6
#define DRAW 507           // 3*13*13
#define DPAD 512           // padded K for WMMA (16 chunks of 32)
#define NTILES (IMW/16)    // 20
#define KCHUNKS (DPAD/32)  // 16

typedef __attribute__((ext_vector_type(16))) _Float16 v16h;
typedef __attribute__((ext_vector_type(8)))  _Float16 v8h;
typedef __attribute__((ext_vector_type(8)))  float    v8f;

// ws layout: lfeat [H][W][512] f16 (A-major, K contiguous)
//            rfeatB [H][32][W][16] f16 (B fragment layout: k = kg*16 + e)
#define LFEAT_HALFS ((size_t)IMH * IMW * DPAD)
#define LFEAT_BYTES (LFEAT_HALFS * 2)

// ---------------------------------------------------------------------------
// Kernel 1: patch z-normalization + fragment-layout scatter.
// One thread per (image, pixel). Two passes over the 507 patch values:
// pass 1 -> mean & norm, pass 2 -> normalized f16 written in WMMA-ready layout.
// Image data (737 KB x2) is L1/L2 resident, so the 2x pass is cheap.
// ---------------------------------------------------------------------------
__global__ void __launch_bounds__(256)
feat_kernel(const float* __restrict__ limg, const float* __restrict__ rimg,
            _Float16* __restrict__ lfeat, _Float16* __restrict__ rfeatB) {
    const int HW = IMH * IMW;
    int idx = blockIdx.x * blockDim.x + threadIdx.x;
    if (idx >= 2 * HW) return;
    const int img = idx / HW;
    const int p   = idx - img * HW;
    const int h   = p / IMW;
    const int w   = p - h * IMW;
    const float* __restrict__ src = img ? rimg : limg;

    // pass 1: stats over zero-padded 13x13x3 patch
    float sum = 0.f, sumsq = 0.f;
    for (int c = 0; c < IMC; ++c) {
        const float* cp = src + (size_t)c * HW;
        for (int i = 0; i < KS; ++i) {
            const int y = h + i - PADK;
            const bool yok = (y >= 0) & (y < IMH);
            const float* rp = cp + (size_t)y * IMW;
            for (int j = 0; j < KS; ++j) {
                const int x = w + j - PADK;
                float v = (yok && x >= 0 && x < IMW) ? rp[x] : 0.f;
                sum += v; sumsq += v * v;
            }
        }
    }
    const float mean = sum * (1.0f / (float)DRAW);
    float var = sumsq - sum * sum * (1.0f / (float)DRAW);
    var = fmaxf(var, 0.f);
    const float inv = 1.0f / (sqrtf(var) + 1e-5f);

    // pass 2: normalized values -> fragment layouts
    int k = 0;
    for (int c = 0; c < IMC; ++c) {
        const float* cp = src + (size_t)c * HW;
        for (int i = 0; i < KS; ++i) {
            const int y = h + i - PADK;
            const bool yok = (y >= 0) & (y < IMH);
            const float* rp = cp + (size_t)y * IMW;
            for (int j = 0; j < KS; ++j) {
                const int x = w + j - PADK;
                float v = (yok && x >= 0 && x < IMW) ? rp[x] : 0.f;
                const _Float16 f = (_Float16)((v - mean) * inv);
                if (img == 0) {
                    lfeat[(size_t)p * DPAD + k] = f;
                } else {
                    // [h][k>>4][w][k&15]
                    rfeatB[(((size_t)h * 32 + (k >> 4)) * IMW + w) * 16 + (k & 15)] = f;
                }
                ++k;
            }
        }
    }
    // zero-fill K padding (507..511) — ws is poisoned, must clear
    for (k = DRAW; k < DPAD; ++k) {
        if (img == 0) lfeat[(size_t)p * DPAD + k] = (_Float16)0.f;
        else rfeatB[(((size_t)h * 32 + (k >> 4)) * IMW + w) * 16 + (k & 15)] = (_Float16)0.f;
    }
}

// ---------------------------------------------------------------------------
// Kernel 2: per-row 320x512x320 correlation via v_wmma_f32_16x16x32_f16,
// fused running argmax + depth. One block per row; 320 threads = 10 waves;
// each wave owns two 16-row M tiles and sweeps all 20 N tiles.
// ---------------------------------------------------------------------------
__global__ void __launch_bounds__(320)
match_kernel(const _Float16* __restrict__ lfeat, const _Float16* __restrict__ rfeatB,
             const float* __restrict__ intri, const float* __restrict__ lext,
             const float* __restrict__ rext, float* __restrict__ out) {
    const int h    = blockIdx.x;
    const int tid  = threadIdx.x;
    const int wave = tid >> 5;
    const int lane = tid & 31;
    const int l16  = lane & 15;
    const bool hihalf = (lane >= 16);

    // camera geometry: rel = r_extri @ inv(l_extri), rigid-transform inverse
    const float fx = intri[0];
    float invt[3];
    #pragma unroll
    for (int i = 0; i < 3; ++i)
        invt[i] = -(lext[0 * 4 + i] * lext[3] + lext[1 * 4 + i] * lext[7] +
                    lext[2 * 4 + i] * lext[11]);
    float bl2 = 0.f;
    #pragma unroll
    for (int i = 0; i < 3; ++i) {
        float ti = rext[i * 4 + 0] * invt[0] + rext[i * 4 + 1] * invt[1] +
                   rext[i * 4 + 2] * invt[2] + rext[i * 4 + 3];
        bl2 += ti * ti;
    }
    const float fxb = fx * sqrtf(bl2);

    // A rows for this wave's two M tiles (ISA A layout: row M = lane%16)
    const int m0 = wave * 32 + l16;
    const _Float16* __restrict__ A0 = lfeat + ((size_t)h * IMW + m0) * DPAD;
    const _Float16* __restrict__ A1 = A0 + (size_t)16 * DPAD;
    const _Float16* __restrict__ Brow = rfeatB + (size_t)h * 32 * IMW * 16;
    const int s = hihalf ? 16 : 0;      // A per-lane K split
    const int kg_off = hihalf ? 1 : 0;  // B per-lane K-group split

    float bestv0[8], bestv1[8];
    int   besti0[8], besti1[8];
    #pragma unroll
    for (int j = 0; j < 8; ++j) {
        bestv0[j] = -3.4e38f; bestv1[j] = -3.4e38f;
        besti0[j] = 0;        besti1[j] = 0;
    }

    for (int nt = 0; nt < NTILES; ++nt) {
        const int nb = nt * 16;
        v8f c0 = {0.f,0.f,0.f,0.f,0.f,0.f,0.f,0.f};
        v8f c1 = {0.f,0.f,0.f,0.f,0.f,0.f,0.f,0.f};
        const _Float16* __restrict__ Bp =
            Brow + ((size_t)kg_off * IMW + nb + l16) * 16;

        #pragma unroll 4
        for (int kc = 0; kc < KCHUNKS; ++kc) {
            const int kb = kc * 32;
            // B fragment: 16 contiguous halfs = one 32B load
            v16h b = *(const v16h*)(Bp + (size_t)(kc * 2) * (IMW * 16));
            // A fragments: two 16B loads each, contiguous k = 16*half + e, as in B
            v8h lo0 = *(const v8h*)(A0 + kb + s);
            v8h hi0 = *(const v8h*)(A0 + kb + 8 + s);
            v8h lo1 = *(const v8h*)(A1 + kb + s);
            v8h hi1 = *(const v8h*)(A1 + kb + 8 + s);
            v16h a0 = __builtin_shufflevector(lo0, hi0, 0,1,2,3,4,5,6,7,8,9,10,11,12,13,14,15);
            v16h a1 = __builtin_shufflevector(lo1, hi1, 0,1,2,3,4,5,6,7,8,9,10,11,12,13,14,15);
            c0 = __builtin_amdgcn_wmma_f32_16x16x32_f16(false, a0, false, b,
                                                        (short)0, c0, false, false);
            c1 = __builtin_amdgcn_wmma_f32_16x16x32_f16(false, a1, false, b,
                                                        (short)0, c1, false, false);
        }
        // running argmax: C layout -> lane holds column v = nb + l16, rows j(+8)
        const int vcol = nb + l16;
        #pragma unroll
        for (int j = 0; j < 8; ++j) {
            if (c0[j] > bestv0[j]) { bestv0[j] = c0[j]; besti0[j] = vcol; }
            if (c1[j] > bestv1[j]) { bestv1[j] = c1[j]; besti1[j] = vcol; }
        }
    }

    // cross-lane argmax over the 16 candidate columns (width-16 butterfly),
    // tie-break toward smaller index to match jnp.argmax first-max semantics
    #pragma unroll
    for (int off = 1; off < 16; off <<= 1) {
        #pragma unroll
        for (int j = 0; j < 8; ++j) {
            float ov = __shfl_xor(bestv0[j], off, 16);
            int   oi = __shfl_xor(besti0[j], off, 16);
            if (ov > bestv0[j] || (ov == bestv0[j] && oi < besti0[j])) {
                bestv0[j] = ov; besti0[j] = oi;
            }
            ov = __shfl_xor(bestv1[j], off, 16);
            oi = __shfl_xor(besti1[j], off, 16);
            if (ov > bestv1[j] || (ov == bestv1[j] && oi < besti1[j])) {
                bestv1[j] = ov; besti1[j] = oi;
            }
        }
    }

    if (l16 == 0) {
        const int rbase0 = wave * 32 + (hihalf ? 8 : 0);      // tile 0 rows
        const int rbase1 = rbase0 + 16;                       // tile 1 rows
        #pragma unroll
        for (int j = 0; j < 8; ++j) {
            int r0 = rbase0 + j;
            float d0 = fmaxf(fabsf((float)besti0[j] - (float)r0), 0.001f);
            out[(size_t)h * IMW + r0] = fxb / d0;
            int r1 = rbase1 + j;
            float d1 = fmaxf(fabsf((float)besti1[j] - (float)r1), 0.001f);
            out[(size_t)h * IMW + r1] = fxb / d1;
        }
    }
}

// ---------------------------------------------------------------------------
extern "C" void kernel_launch(void* const* d_in, const int* in_sizes, int n_in,
                              void* d_out, int out_size, void* d_ws, size_t ws_size,
                              hipStream_t stream) {
    const float* l_image = (const float*)d_in[0];
    const float* r_image = (const float*)d_in[1];
    const float* l_intri = (const float*)d_in[2];
    const float* l_extri = (const float*)d_in[4];
    const float* r_extri = (const float*)d_in[5];
    float* out = (float*)d_out;

    _Float16* lfeat  = (_Float16*)d_ws;
    _Float16* rfeatB = (_Float16*)((char*)d_ws + LFEAT_BYTES);

    // 1) build z-normalized patch features in WMMA fragment layouts
    const int total = 2 * IMH * IMW;
    feat_kernel<<<(total + 255) / 256, 256, 0, stream>>>(l_image, r_image, lfeat, rfeatB);

    // 2) per-row WMMA correlation + argmax + depth
    match_kernel<<<IMH, 320, 0, stream>>>(lfeat, rfeatB, l_intri, l_extri, r_extri, out);
}
